// Net_171798692309
// MI455X (gfx1250) — matrix operand order repaired and checked by reading, not compiled, not run on hardware
//
#include <hip/hip_runtime.h>

typedef __attribute__((ext_vector_type(16))) __bf16 v16bf;
typedef __attribute__((ext_vector_type(8)))  float  v8f;

#define HID 256

// ---------- weight convert+transpose: W[K,N] f32 -> Wt[N,Kp] bf16 (zero-pad K..Kp)
__global__ void k_wt(const float* __restrict__ W, __bf16* __restrict__ Wt,
                     int K, int N, int Kp, int total) {
  int idx = blockIdx.x * 256 + threadIdx.x;
  if (idx >= total) return;
  int n = idx / Kp, k = idx - n * Kp;
  float v = (k < K) ? W[(size_t)k * N + n] : 0.f;
  Wt[idx] = (__bf16)v;
}

// ---------- degree (incl. self loop) and D^{-1/2}
__global__ void k_deg_init(float* deg, int n) {
  int i = blockIdx.x * 256 + threadIdx.x;
  if (i < n) deg[i] = 1.0f;               // self-loop weight
}
__global__ void k_deg_edges(const int* __restrict__ col, const float* __restrict__ ew,
                            float* deg, int E) {
  int e = blockIdx.x * 256 + threadIdx.x;
  if (e < E) atomicAdd(&deg[col[e]], ew[e]);
}
__global__ void k_dis(const float* __restrict__ deg, float* dis, int n) {
  int i = blockIdx.x * 256 + threadIdx.x;
  if (i < n) { float d = deg[i]; dis[i] = d > 0.f ? rsqrtf(d) : 0.f; }
}

// ---------- bf16 WMMA GEMM: C[M,256] = A[M,K] * B[K,256]
// A: f32 row-major (optional fused ReLU). Bt: bf16 [256][Kp] transposed weights.
// Wave tile: 16(M) x 256(N); WG = 8 waves -> 128 rows/block.
// B double-buffered in LDS via async global->LDS copies (ASYNCcnt).
__global__ __launch_bounds__(256) void k_gemm_bf16(
    const float* __restrict__ A, const __bf16* __restrict__ Bt,
    float* __restrict__ C, int M, int K, int Kp, int relu)
{
  __shared__ __align__(32) __bf16 lb[2][HID * 32];   // 2 x [N=256][K=32] = 32 KB
  const int tid  = threadIdx.x;
  const int wave = tid >> 5, lane = tid & 31;
  const int lrow = lane & 15, lhi = lane >> 4;
  const int m0 = blockIdx.x * 128 + wave * 16;
  int mrow = m0 + lrow; if (mrow > M - 1) mrow = M - 1;   // clamp; stores guarded
  const float* __restrict__ arow = A + (size_t)mrow * K;

  // per-thread async staging addresses: 4 x b128 chunks per thread per tile
  const unsigned long long btp = (unsigned long long)Bt;
  unsigned gbase[4], l0[4], l1[4];
  {
    const unsigned lds0 = (unsigned)(size_t)(&lb[0][0]);
    const unsigned lds1 = (unsigned)(size_t)(&lb[1][0]);
    #pragma unroll
    for (int j = 0; j < 4; ++j) {
      int idx = tid + j * 256;                // 1024 chunks of 8 bf16
      int n = idx >> 2, ko = (idx & 3) << 3;
      gbase[j] = (unsigned)(((size_t)n * Kp + ko) * 2);
      unsigned loff = (unsigned)((n * 32 + ko) * 2);
      l0[j] = lds0 + loff;
      l1[j] = lds1 + loff;
    }
  }
  auto stage = [&](int buf, int k0) {
    #pragma unroll
    for (int j = 0; j < 4; ++j) {
      unsigned ga = gbase[j] + (unsigned)(k0 * 2);
      unsigned la = buf ? l1[j] : l0[j];
      asm volatile("global_load_async_to_lds_b128 %0, %1, %2 offset:0"
                   :: "v"(la), "v"(ga), "s"(btp) : "memory");
    }
  };

  v8f acc[16];
  #pragma unroll
  for (int t = 0; t < 16; ++t)
    #pragma unroll
    for (int i = 0; i < 8; ++i) acc[t][i] = 0.f;

  const int ksteps = Kp >> 5;
  stage(0, 0);                                // prologue: stage tile 0
  for (int ks = 0; ks < ksteps; ++ks) {
    const int k0 = ks << 5;
    const int buf = ks & 1;
    if (ks + 1 < ksteps) {
      stage(buf ^ 1, k0 + 32);                // stage next tile while computing
      asm volatile("s_wait_asynccnt 0x4" ::: "memory");   // oldest stage done
    } else {
      asm volatile("s_wait_asynccnt 0x0" ::: "memory");
    }
    __syncthreads();                          // all waves' stage visible

    // A fragment: lane holds row M=lrow, K c..c+15, c = lhi*16
    const int ka = k0 + lhi * 16;
    float av[16];
    if (k0 + 32 <= K) {
      float4 a0 = *(const float4*)(arow + ka);
      float4 a1 = *(const float4*)(arow + ka + 4);
      float4 a2 = *(const float4*)(arow + ka + 8);
      float4 a3 = *(const float4*)(arow + ka + 12);
      av[0]=a0.x; av[1]=a0.y; av[2]=a0.z; av[3]=a0.w;
      av[4]=a1.x; av[5]=a1.y; av[6]=a1.z; av[7]=a1.w;
      av[8]=a2.x; av[9]=a2.y; av[10]=a2.z; av[11]=a2.w;
      av[12]=a3.x; av[13]=a3.y; av[14]=a3.z; av[15]=a3.w;
    } else {                                  // K tail (K=1700): zero fill
      #pragma unroll
      for (int i = 0; i < 16; ++i) {
        int kk = ka + i;
        av[i] = (kk < K) ? arow[kk] : 0.f;
      }
    }
    if (relu) {
      #pragma unroll
      for (int i = 0; i < 16; ++i) av[i] = fmaxf(av[i], 0.f);
    }
    v16bf af;
    #pragma unroll
    for (int i = 0; i < 16; ++i) af[i] = (__bf16)av[i];

    if (ks + 1 < ksteps) __builtin_prefetch(arow + ka + 32, 0, 1);

    #pragma unroll
    for (int nt = 0; nt < 16; ++nt) {
      // B fragment: lane = column N = nt*16+lrow, K 0-15 (lanes 0-15) / 16-31 (lanes 16-31)
      const v16bf bf = *(const v16bf*)(&lb[buf][(nt * 16 + lrow) * 32 + lhi * 16]);
      acc[nt] = __builtin_amdgcn_wmma_f32_16x16x32_bf16(
          false, af, false, bf, (short)0, acc[nt], false, false);
    }
    __syncthreads();                          // done reading buf before re-stage
  }

  // C/D layout: VGPR r, lane L -> M = r + 8*(L>=16), N = L&15
  #pragma unroll
  for (int nt = 0; nt < 16; ++nt) {
    #pragma unroll
    for (int r = 0; r < 8; ++r) {
      int m = m0 + r + lhi * 8;
      if (m < M) C[(size_t)m * HID + nt * 16 + lrow] = acc[nt][r];
    }
  }
}

// ---------- aggregation: out = selfnorm*H + b  (init), then += norm_e * H[row]
__global__ void k_agg_init(const float* __restrict__ H, const float* __restrict__ dis,
                           const float* __restrict__ b, float* __restrict__ out, int n) {
  int idx = blockIdx.x * 256 + threadIdx.x;
  if (idx >= n * HID) return;
  int i = idx >> 8, f = idx & 255;
  float s = dis[i];
  out[idx] = s * s * H[idx] + b[f];
}

__global__ __launch_bounds__(256) void k_agg_edges(
    const float* __restrict__ H, const int* __restrict__ row, const int* __restrict__ col,
    const float* __restrict__ ew, const float* __restrict__ dis,
    float* __restrict__ out, int E) {
  int e = blockIdx.x * 4 + (threadIdx.x >> 6);   // 64 lanes per edge, 4 floats each
  if (e >= E) return;
  int t = (threadIdx.x & 63) << 2;
  int r = row[e], c = col[e];
  float nm = dis[r] * ew[e] * dis[c];
  const float4 h = *(const float4*)(H + (size_t)r * HID + t);
  float* o = out + (size_t)c * HID + t;
  atomicAdd(o + 0, nm * h.x);
  atomicAdd(o + 1, nm * h.y);
  atomicAdd(o + 2, nm * h.z);
  atomicAdd(o + 3, nm * h.w);
}

// ---------- mean pool + final linear
__global__ void k_zero(float* g, int n) {
  int i = blockIdx.x * 256 + threadIdx.x;
  if (i < n) g[i] = 0.f;
}
__global__ void k_pool(const float* __restrict__ H, float* __restrict__ g, int n) {
  float acc = 0.f;
  int f = threadIdx.x;
  for (int i = blockIdx.x; i < n; i += gridDim.x)
    acc += H[(size_t)i * HID + f];
  atomicAdd(&g[f], acc);
}
__global__ void k_final(const float* __restrict__ g, const float* __restrict__ Wl,
                        const float* __restrict__ bl, float* __restrict__ out,
                        int F, float invn) {
  __shared__ float gs[HID];
  if (threadIdx.x < HID) gs[threadIdx.x] = g[threadIdx.x] * invn;
  __syncthreads();
  int j = blockIdx.x * 256 + threadIdx.x;
  if (j >= F) return;
  float acc = bl[j];
  for (int k = 0; k < HID; ++k) acc += gs[k] * Wl[(size_t)k * F + j];
  out[j] = acc;
}

extern "C" void kernel_launch(void* const* d_in, const int* in_sizes, int n_in,
                              void* d_out, int out_size, void* d_ws, size_t ws_size,
                              hipStream_t stream) {
  const float* x  = (const float*)d_in[0];
  const int*   ei = (const int*)d_in[1];
  const float* ea = (const float*)d_in[2];
  const float* W1 = (const float*)d_in[3]; const float* b1 = (const float*)d_in[4];
  const float* W2 = (const float*)d_in[5]; const float* b2 = (const float*)d_in[6];
  const float* W3 = (const float*)d_in[7]; const float* b3 = (const float*)d_in[8];
  const float* Wl = (const float*)d_in[9]; const float* bl = (const float*)d_in[10];
  float* out = (float*)d_out;

  const int F   = in_sizes[10];           // 1700
  const int N   = in_sizes[0] / F;        // 20000
  const int E   = in_sizes[2];            // 320000
  const int Kp1 = (F + 31) & ~31;         // 1728 (rows stay 16B-aligned in bf16)

  const int* row = ei;
  const int* col = ei + E;

  char* w = (char*)d_ws;
  auto carve = [&](size_t bytes) { char* p = w; w += (bytes + 255) & ~(size_t)255; return p; };
  float*  H   = (float*) carve((size_t)N * HID * 4);
  float*  G   = (float*) carve((size_t)N * HID * 4);
  float*  deg = (float*) carve((size_t)N * 4);
  float*  dis = (float*) carve((size_t)N * 4);
  float*  gp  = (float*) carve(HID * 4);
  __bf16* Wt1 = (__bf16*)carve((size_t)HID * Kp1 * 2);
  __bf16* Wt2 = (__bf16*)carve((size_t)HID * HID * 2);
  __bf16* Wt3 = (__bf16*)carve((size_t)HID * HID * 2);

  { int tot = HID * Kp1; k_wt<<<(tot + 255)/256, 256, 0, stream>>>(W1, Wt1, F,   HID, Kp1, tot); }
  { int tot = HID * HID; k_wt<<<(tot + 255)/256, 256, 0, stream>>>(W2, Wt2, HID, HID, HID, tot); }
  { int tot = HID * HID; k_wt<<<(tot + 255)/256, 256, 0, stream>>>(W3, Wt3, HID, HID, HID, tot); }

  k_deg_init <<<(N + 255)/256, 256, 0, stream>>>(deg, N);
  k_deg_edges<<<(E + 255)/256, 256, 0, stream>>>(col, ea, deg, E);
  k_dis      <<<(N + 255)/256, 256, 0, stream>>>(deg, dis, N);

  const int gblk = (N + 127) / 128;
  const int ntot = N * HID;
  const int eblk = (E + 3) / 4;

  // layer 1
  k_gemm_bf16<<<gblk, 256, 0, stream>>>(x, Wt1, H, N, F, Kp1, 0);
  k_agg_init <<<(ntot + 255)/256, 256, 0, stream>>>(H, dis, b1, G, N);
  k_agg_edges<<<eblk, 256, 0, stream>>>(H, row, col, ea, dis, G, E);
  // layer 2 (ReLU fused on A read)
  k_gemm_bf16<<<gblk, 256, 0, stream>>>(G, Wt2, H, N, HID, HID, 1);
  k_agg_init <<<(ntot + 255)/256, 256, 0, stream>>>(H, dis, b2, G, N);
  k_agg_edges<<<eblk, 256, 0, stream>>>(H, row, col, ea, dis, G, E);
  // layer 3 (ReLU fused on A read; no ReLU after)
  k_gemm_bf16<<<gblk, 256, 0, stream>>>(G, Wt3, H, N, HID, HID, 1);
  k_agg_init <<<(ntot + 255)/256, 256, 0, stream>>>(H, dis, b3, G, N);
  k_agg_edges<<<eblk, 256, 0, stream>>>(H, row, col, ea, dis, G, E);
  // pool + final linear
  k_zero <<<1, 256, 0, stream>>>(gp, HID);
  k_pool <<<120, 256, 0, stream>>>(G, gp, N);
  k_final<<<(F + 255)/256, 256, 0, stream>>>(gp, Wl, bl, out, F, 1.0f / (float)N);
}
